// DepthDeformConv_54082228191689
// MI455X (gfx1250) — matrix operand order repaired and checked by reading, not compiled, not run on hardware
//
#include <hip/hip_runtime.h>
#include <hip/hip_bf16.h>

// ---------------------------------------------------------------------------
// DepthDeformConv for MI455X (gfx1250, wave32, WMMA).
//
// Reference: DCNv2 deformable conv, B=4, C=64, H=W=128, O=64, 3x3, s=1, p=1.
// (`depth` input is unused by the reference.)
// Strategy:
//   1) weight_to_f16: one-shot f32 -> f16 convert of weights into d_ws
//      (O x 576 contiguous, 72 KB, L2-resident, reused by all blocks).
//   2) deform_conv_wmma: per block (256 thr = 8 waves), one 32-pixel tile.
//      Phase 1: bilinear im2col -> f16 cols tile in LDS (36.5 KB, padded).
//      Phase 2: 8 waves = 4 (M=16 ch) x 2 (N=16 pix) tiles, 18 K-steps of 32
//               using v_wmma_f32_16x16x32_f16 with f32 accumulation.
// ---------------------------------------------------------------------------

typedef __attribute__((ext_vector_type(16))) _Float16 v16h;
typedef __attribute__((ext_vector_type(8)))  _Float16 v8h;
typedef __attribute__((ext_vector_type(8)))  float    v8f;

#define BATCH    4
#define C_IN     64
#define HH       128
#define WW       128
#define HW       (HH * WW)          // 16384
#define O_OUT    64
#define KTAPS    9
#define KDIM     (C_IN * KTAPS)     // 576 = GEMM K
#define KPAD     (KDIM + 8)         // 584 halves: 1168B pixel stride, 16B aligned,
                                    // 292-dword LDS stride -> spread banks
#define TILE_PIX 32
#define KSTEPS   (KDIM / 32)        // 18 WMMA K-steps

__global__ void weight_to_f16(const float* __restrict__ w,
                              _Float16* __restrict__ wh, int n) {
  int i = blockIdx.x * blockDim.x + threadIdx.x;
  if (i < n) wh[i] = (_Float16)w[i];
}

__global__ __launch_bounds__(256)
void deform_conv_wmma(const float* __restrict__ input,
                      const float* __restrict__ offset,
                      const float* __restrict__ mask,
                      const _Float16* __restrict__ wgt,   // [O][576] f16
                      const float* __restrict__ bias,
                      float* __restrict__ out) {
  __shared__ _Float16 cols[TILE_PIX * KPAD];   // 37376 bytes

  const int lane    = threadIdx.x & 31;
  const int wv      = threadIdx.x >> 5;        // 0..7
  const int laneM   = lane & 15;
  const int laneHi  = lane >> 4;               // 0/1 (K-half selector)
  const int pixBase = blockIdx.x * TILE_PIX;   // 32 consecutive pixels, one row

  const int mIdx = wv & 3;                     // which 16-channel output block
  const int nh   = wv >> 2;                    // which 16-pixel half of the tile

  // Stage this wave's weight rows into cache while phase 1 samples.
  // Row = 1152 B = 9 x 128 B; laneHi parity covers all 9 segments per row pair.
  {
    const _Float16* wrow = wgt + (mIdx * 16 + laneM) * KDIM;
    #pragma unroll
    for (int j = 0; j < 9; ++j)
      if ((j & 1) == laneHi) __builtin_prefetch(wrow + j * 64, 0, 0);
  }

  // -------- Phase 1: bilinear im2col into LDS (f16) --------
  {
    const int pix = lane;                      // lane <-> pixel in tile
    const int p   = pixBase + pix;
    const int b   = p >> 14;                   // /16384
    const int hw  = p & (HW - 1);
    const int h   = hw >> 7;
    const int w   = hw & (WW - 1);
    const float* inB  = input  + (size_t)b * C_IN * HW;
    const float* offB = offset + (size_t)b * 2 * KTAPS * HW + hw;
    const float* mskB = mask   + (size_t)b * KTAPS * HW + hw;
    const int cg = wv;                         // 8-channel group per wave

    #pragma unroll
    for (int k = 0; k < KTAPS; ++k) {
      const float oy   = offB[(2 * k) * HW];
      const float ox   = offB[(2 * k + 1) * HW];
      const float mval = mskB[k * HW];

      const float py = (float)(h - 1 + k / 3) + oy;
      const float px = (float)(w - 1 + k % 3) + ox;
      const float fy = floorf(py), fx = floorf(px);
      const int y0 = (int)fy, x0 = (int)fx;
      const int y1 = y0 + 1,  x1 = x0 + 1;
      const float wy = py - fy, wx = px - fx;

      const float vy0 = (y0 >= 0 && y0 < HH) ? 1.0f : 0.0f;
      const float vy1 = (y1 >= 0 && y1 < HH) ? 1.0f : 0.0f;
      const float vx0 = (x0 >= 0 && x0 < WW) ? 1.0f : 0.0f;
      const float vx1 = (x1 >= 0 && x1 < WW) ? 1.0f : 0.0f;

      const int yc0 = min(max(y0, 0), HH - 1);
      const int yc1 = min(max(y1, 0), HH - 1);
      const int xc0 = min(max(x0, 0), WW - 1);
      const int xc1 = min(max(x1, 0), WW - 1);

      // Fold mask into the bilinear weights.
      const float w00 = (1.0f - wy) * (1.0f - wx) * vy0 * vx0 * mval;
      const float w01 = (1.0f - wy) * wx          * vy0 * vx1 * mval;
      const float w10 = wy          * (1.0f - wx) * vy1 * vx0 * mval;
      const float w11 = wy          * wx          * vy1 * vx1 * mval;

      const int i00 = yc0 * WW + xc0;
      const int i01 = yc0 * WW + xc1;
      const int i10 = yc1 * WW + xc0;
      const int i11 = yc1 * WW + xc1;

      const float* ch = inB + (size_t)(cg * 8) * HW;
      _Float16* dst = &cols[pix * KPAD + k];   // + c*9 below
      #pragma unroll
      for (int cc = 0; cc < 8; ++cc) {
        const float v = w00 * ch[i00] + w01 * ch[i01] +
                        w10 * ch[i10] + w11 * ch[i11];
        dst[(cg * 8 + cc) * KTAPS] = (_Float16)v;
        ch += HW;
      }
    }
  }

  __syncthreads();

  // -------- Phase 2: WMMA GEMM  out[16 ch x 16 pix] per wave --------
  // A (16x32 f16, ISA layout): lane M = laneM; K chunks 8*laneHi and 16+8*laneHi.
  // B (32x16 f16):            lane N = laneM; K chunks 8*laneHi and 16+8*laneHi.
  const int o = mIdx * 16 + laneM;
  const _Float16* arow = wgt + (size_t)o * KDIM;
  const _Float16* brow = &cols[(nh * 16 + laneM) * KPAD];

  v8f acc = {};
  for (int kk = 0; kk < KSTEPS; ++kk) {
    const int kb = kk * 32;
    union { v16h v; v8h h[2]; } A, B;
    A.h[0] = *(const v8h*)(arow + kb + laneHi * 8);        // K = kb + {0..7|8..15}
    A.h[1] = *(const v8h*)(arow + kb + 16 + laneHi * 8);   // K = kb + 16 + ...
    B.h[0] = *(const v8h*)(brow + kb + laneHi * 8);        // K = kb + 8*laneHi + 0..7
    B.h[1] = *(const v8h*)(brow + kb + 16 + laneHi * 8);   // K = kb + 16 + 8*laneHi + 0..7
    acc = __builtin_amdgcn_wmma_f32_16x16x32_f16(
        /*neg_a=*/false, A.v, /*neg_b=*/false, B.v,
        /*c_mod=*/(short)0, acc, /*reuse_a=*/false, /*reuse_b=*/false);
  }

  // -------- Epilogue: D layout -> global (coalesced f32) --------
  // VGPR v: M = v + 8*laneHi, N = laneM.
  const int pg  = pixBase + nh * 16 + laneM;   // global pixel of this lane
  const int bb  = pg >> 14;
  const int hwv = pg & (HW - 1);
  float* outB = out + (size_t)bb * O_OUT * HW + hwv;
  #pragma unroll
  for (int v = 0; v < 8; ++v) {
    const int oc = mIdx * 16 + v + 8 * laneHi;
    outB[(size_t)oc * HW] = acc[v] + bias[oc];
  }
}

extern "C" void kernel_launch(void* const* d_in, const int* in_sizes, int n_in,
                              void* d_out, int out_size, void* d_ws, size_t ws_size,
                              hipStream_t stream) {
  const float* input  = (const float*)d_in[0];
  // d_in[1] = depth: unused by the reference computation.
  const float* offset = (const float*)d_in[2];
  const float* mask   = (const float*)d_in[3];
  const float* weight = (const float*)d_in[4];
  const float* bias   = (const float*)d_in[5];
  float* out = (float*)d_out;

  _Float16* wh = (_Float16*)d_ws;              // needs 64*576*2 = 73728 B
  const int nW = O_OUT * KDIM;
  weight_to_f16<<<(nW + 255) / 256, 256, 0, stream>>>(weight, wh, nW);

  const int tiles = (BATCH * HW) / TILE_PIX;   // 2048 blocks
  deform_conv_wmma<<<tiles, 256, 0, stream>>>(input, offset, mask, wh, bias, out);
}
